// RGTLayer_59176059404896
// MI455X (gfx1250) — matrix operand order repaired and checked by reading, not compiled, not run on hardware
//
#include <hip/hip_runtime.h>
#include <hip/hip_bf16.h>
#include <math.h>

#define N_NODES 50000
#define N_EDGES 500000
#define QKV_W   896   // 256 q | 256 k | 256 v | 128 skip

typedef __attribute__((ext_vector_type(16))) __bf16         v16bf;
typedef __attribute__((ext_vector_type(16))) unsigned short v16us;
typedef __attribute__((ext_vector_type(8)))  unsigned short v8us;
typedef __attribute__((ext_vector_type(4)))  unsigned short v4us;
typedef __attribute__((ext_vector_type(8)))  float          v8f;
typedef __attribute__((ext_vector_type(4)))  float          v4f;

union V16 { v16us v; v8us h[2]; };

__device__ __forceinline__ unsigned short f2bf(float f) {
  union { float f; unsigned u; } x; x.f = f;
  unsigned r = x.u + 0x7FFFu + ((x.u >> 16) & 1u);   // round-to-nearest-even
  return (unsigned short)(r >> 16);
}

// Accumulate `ksteps` chunks of K=32 into acc via v_wmma_f32_16x16x32_bf16.
// aRow/bRow already include this lane's (row*ld + hi*8) offset.
// A layout (16-bit, 16x32): lane<16 -> M=lane, K=j(+8 hi-j); lane>=16 shifts K by 8.
// B layout (col-major staged Wt): lane holds column n0+(lane&15), same K order as A.
__device__ __forceinline__ void wmma_chain(v8f& acc,
                                           const unsigned short* __restrict__ aRow,
                                           const unsigned short* __restrict__ bRow,
                                           int ksteps) {
#pragma unroll
  for (int t = 0; t < ksteps; ++t) {
    V16 ta, tb;
    ta.h[0] = *(const v8us*)(aRow + t * 32);
    ta.h[1] = *(const v8us*)(aRow + t * 32 + 16);
    tb.h[0] = *(const v8us*)(bRow + t * 32);
    tb.h[1] = *(const v8us*)(bRow + t * 32 + 16);
    acc = __builtin_amdgcn_wmma_f32_16x16x32_bf16(
        false, __builtin_bit_cast(v16bf, ta.v),
        false, __builtin_bit_cast(v16bf, tb.v),
        (short)0, acc, false, false);
  }
}

// ---------------- prep kernels ----------------

// 4 elements per thread, b128 in / b64 out
__global__ void k_f32_to_bf16(const float* __restrict__ X,
                              unsigned short* __restrict__ Y, long n4) {
  long i = (long)blockIdx.x * blockDim.x + threadIdx.x;
  if (i >= n4) return;
  v4f x = *(const v4f*)(X + i * 4);
  v4us y;
#pragma unroll
  for (int j = 0; j < 4; ++j) y[j] = f2bf(x[j]);
  *(v4us*)(Y + i * 4) = y;
}

// W is [K][NC] row-major f32; write Wt[(colOff+col)][k] bf16 (leading dim outLd)
__global__ void k_transpose_cvt(const float* __restrict__ W,
                                unsigned short* __restrict__ Wt,
                                int K, int NC, int outLd, int colOff) {
  int idx = blockIdx.x * blockDim.x + threadIdx.x;
  if (idx >= K * NC) return;
  int col = idx / K, k = idx % K;
  Wt[(size_t)(colOff + col) * outLd + k] = f2bf(W[(size_t)k * NC + col]);
}

// ---------------- fused q/k/v/skip GEMM: C[N,896] = xbf @ W + bias ----------------

__global__ void k_gemm_qkvs(const unsigned short* __restrict__ xbf,
                            const unsigned short* __restrict__ Wt,
                            const float* __restrict__ bias,
                            float* __restrict__ C) {
  int lane = threadIdx.x & 31, wv = threadIdx.x >> 5;
  int n0 = (blockIdx.x * 8 + wv) * 16;          // 56 column tiles, 7 blocks * 8 waves
  int m0 = blockIdx.y * 16;
  int row = lane & 15, hi = lane >> 4;
  v8f acc = {0.f, 0.f, 0.f, 0.f, 0.f, 0.f, 0.f, 0.f};
  wmma_chain(acc, xbf + (size_t)(m0 + row) * 128 + hi * 8,
                  Wt  + (size_t)(n0 + row) * 128 + hi * 8, 4);
  float b = bias[n0 + row];
#pragma unroll
  for (int i = 0; i < 8; ++i)
    C[(size_t)(m0 + hi * 8 + i) * QKV_W + n0 + row] = acc[i] + b;
}

// ---------------- edge pass 1: logits -> exp -> segment sum ----------------
// one wave per edge; softmax-shift invariance lets us skip the segment max.
// lane handles 4 contiguous channels -> one b128 gather per head per operand.

__global__ void k_edge_attn(const int* __restrict__ src, const int* __restrict__ dst,
                            const float* __restrict__ qkvs,
                            float* __restrict__ expv, float* __restrict__ ssum) {
  int lane = threadIdx.x & 31;
  int e = blockIdx.x * 8 + (threadIdx.x >> 5);
  if (e >= N_EDGES) return;
  int s = src[e], d = dst[e];
  const v4f* q = (const v4f*)(qkvs + (size_t)d * QKV_W);          // cols 0..255
  const v4f* k = (const v4f*)(qkvs + (size_t)s * QKV_W + 256);    // cols 256..511
  const float scale = 0.08838834764831845f;                       // 1/sqrt(128)
#pragma unroll
  for (int h = 0; h < 2; ++h) {
    v4f qv = q[h * 32 + lane];
    v4f kv = k[h * 32 + lane];
    float p = qv[0] * kv[0] + qv[1] * kv[1] + qv[2] * kv[2] + qv[3] * kv[3];
    p += __shfl_xor(p, 16, 32);
    p += __shfl_xor(p, 8, 32);
    p += __shfl_xor(p, 4, 32);
    p += __shfl_xor(p, 2, 32);
    p += __shfl_xor(p, 1, 32);
    if (lane == 0) {
      float ev = expf(p * scale);
      expv[(size_t)e * 2 + h] = ev;
      atomicAdd(&ssum[(size_t)d * 2 + h], ev);
    }
  }
}

// ---------------- edge pass 2: scatter alpha * v ----------------

__global__ void k_edge_scatter(const int* __restrict__ src, const int* __restrict__ dst,
                               const float* __restrict__ qkvs,
                               const float* __restrict__ expv,
                               const float* __restrict__ ssum,
                               float* __restrict__ agg) {
  int lane = threadIdx.x & 31;
  int e = blockIdx.x * 8 + (threadIdx.x >> 5);
  if (e >= N_EDGES) return;
  int s = src[e], d = dst[e];
  const v4f* v = (const v4f*)(qkvs + (size_t)s * QKV_W + 512);    // cols 512..767
  float* ag = agg + (size_t)d * 256;
#pragma unroll
  for (int h = 0; h < 2; ++h) {
    float al = expv[(size_t)e * 2 + h] / fmaxf(ssum[(size_t)d * 2 + h], 1e-16f);
    v4f vv = v[h * 32 + lane];
    int cbase = h * 128 + lane * 4;
#pragma unroll
    for (int j = 0; j < 4; ++j)
      atomicAdd(&ag[cbase + j], al * vv[j]);
  }
}

// ---------------- u = mean_heads(agg) + skip (4 channels / thread) ----------------

__global__ void k_make_u(const float* __restrict__ agg, const float* __restrict__ qkvs,
                         float* __restrict__ u, unsigned short* __restrict__ ubf) {
  long i = (long)blockIdx.x * blockDim.x + threadIdx.x;
  if (i >= (long)N_NODES * 32) return;
  long n = i >> 5; int c4 = (int)(i & 31) * 4;
  v4f a0 = *(const v4f*)(agg + n * 256 + c4);
  v4f a1 = *(const v4f*)(agg + n * 256 + 128 + c4);
  v4f sk = *(const v4f*)(qkvs + n * QKV_W + 768 + c4);
  v4f val = 0.5f * (a0 + a1) + sk;
  *(v4f*)(u + n * 128 + c4) = val;
  v4us r;
#pragma unroll
  for (int j = 0; j < 4; ++j) r[j] = f2bf(val[j]);
  *(v4us*)(ubf + n * 128 + c4) = r;
}

// ---------------- gate GEMM ([u|x] @ Wg) fused with gating epilogue ----------------

__global__ void k_gate_fused(const unsigned short* __restrict__ ubf,
                             const unsigned short* __restrict__ xbf,
                             const unsigned short* __restrict__ Wgt,  // [128 cols][256 k]
                             const float* __restrict__ gate_b,
                             const float* __restrict__ u, const float* __restrict__ x,
                             float* __restrict__ z, unsigned short* __restrict__ zbf) {
  int lane = threadIdx.x & 31, wv = threadIdx.x >> 5;
  int n0 = wv * 16;                 // 8 column tiles per block
  int m0 = blockIdx.x * 16;
  int row = lane & 15, hi = lane >> 4;
  v8f acc = {0.f, 0.f, 0.f, 0.f, 0.f, 0.f, 0.f, 0.f};
  // rows 0..127 of gate_W multiply u; rows 128..255 multiply x
  wmma_chain(acc, ubf + (size_t)(m0 + row) * 128 + hi * 8,
                  Wgt + (size_t)(n0 + row) * 256 + hi * 8, 4);
  wmma_chain(acc, xbf + (size_t)(m0 + row) * 128 + hi * 8,
                  Wgt + (size_t)(n0 + row) * 256 + 128 + hi * 8, 4);
  float gb = gate_b[n0 + row];
#pragma unroll
  for (int i = 0; i < 8; ++i) {
    int n = m0 + hi * 8 + i, c = n0 + row;
    float g  = acc[i] + gb;
    float a  = 1.f / (1.f + expf(-g));
    float uu = u[(size_t)n * 128 + c];
    float xx = x[(size_t)n * 128 + c];
    float zz = tanhf(uu) * a + xx * (1.f - a);
    z[(size_t)n * 128 + c]   = zz;
    zbf[(size_t)n * 128 + c] = f2bf(zz);
  }
}

// ---------------- semantic head: sum_n sum_c tanh(z@W1 + b1)*w2 ----------------

__global__ void k_sem_reduce(const unsigned short* __restrict__ zbf,
                             const unsigned short* __restrict__ W1t,
                             const float* __restrict__ b1, const float* __restrict__ w2,
                             float* __restrict__ wslot) {
  int lane = threadIdx.x & 31, wv = threadIdx.x >> 5;
  int n0 = wv * 16;
  int m0 = blockIdx.x * 16;
  int row = lane & 15, hi = lane >> 4;
  v8f acc = {0.f, 0.f, 0.f, 0.f, 0.f, 0.f, 0.f, 0.f};
  wmma_chain(acc, zbf + (size_t)(m0 + row) * 128 + hi * 8,
                  W1t + (size_t)(n0 + row) * 128 + hi * 8, 4);
  float bb = b1[n0 + row], ww = w2[n0 + row];
  float local = 0.f;
#pragma unroll
  for (int i = 0; i < 8; ++i) local += tanhf(acc[i] + bb) * ww;
  local += __shfl_xor(local, 16, 32);
  local += __shfl_xor(local, 8, 32);
  local += __shfl_xor(local, 4, 32);
  local += __shfl_xor(local, 2, 32);
  local += __shfl_xor(local, 1, 32);
  if (lane == 0) atomicAdd(wslot, local);
}

__global__ void k_sem_coef(const float* __restrict__ wsum, float* __restrict__ coef) {
  if (threadIdx.x == 0 && blockIdx.x == 0) {
    float c0 = 0.f, c1 = 0.f;
    for (int h = 0; h < 2; ++h) {
      float m0 = wsum[h * 2 + 0] / (float)N_NODES;
      float m1 = wsum[h * 2 + 1] / (float)N_NODES;
      float mx = fmaxf(m0, m1);
      float e0 = expf(m0 - mx), e1 = expf(m1 - mx);
      float s = e0 + e1;
      c0 += e0 / s; c1 += e1 / s;
    }
    coef[0] = c0 * 0.5f;   // already includes 1/SEM_HEADS
    coef[1] = c1 * 0.5f;
  }
}

__global__ void k_final(const float* __restrict__ z0, const float* __restrict__ z1,
                        const float* __restrict__ coef, float* __restrict__ out) {
  long i = (long)blockIdx.x * blockDim.x + threadIdx.x;
  if (i >= (long)N_NODES * 32) return;
  float c0 = coef[0], c1 = coef[1];
  v4f a = *(const v4f*)(z0 + i * 4);
  v4f b = *(const v4f*)(z1 + i * 4);
  *(v4f*)(out + i * 4) = c0 * a + c1 * b;
}

// ---------------- host ----------------

extern "C" void kernel_launch(void* const* d_in, const int* in_sizes, int n_in,
                              void* d_out, int out_size, void* d_ws, size_t ws_size,
                              hipStream_t stream) {
  (void)in_sizes; (void)n_in; (void)out_size; (void)ws_size;

  const float* features = (const float*)d_in[0];
  const int* ei[2] = {(const int*)d_in[1], (const int*)d_in[2]};
  const float* rW[2][4]; const float* rb[2][4];
  for (int r = 0; r < 2; ++r) {
    int b = 3 + r * 8;
    rW[r][0] = (const float*)d_in[b + 0]; rb[r][0] = (const float*)d_in[b + 1]; // Wq,bq
    rW[r][1] = (const float*)d_in[b + 2]; rb[r][1] = (const float*)d_in[b + 3]; // Wk,bk
    rW[r][2] = (const float*)d_in[b + 4]; rb[r][2] = (const float*)d_in[b + 5]; // Wv,bv
    rW[r][3] = (const float*)d_in[b + 6]; rb[r][3] = (const float*)d_in[b + 7]; // Ws,bs
  }
  const float* gate_W = (const float*)d_in[19];
  const float* gate_b = (const float*)d_in[20];
  const float* sW1[2] = {(const float*)d_in[21], (const float*)d_in[24]};
  const float* sb1[2] = {(const float*)d_in[22], (const float*)d_in[25]};
  const float* sw2[2] = {(const float*)d_in[23], (const float*)d_in[26]};
  float* out = (float*)d_out;

  char* wb = (char*)d_ws;
  size_t off = 0;
  auto carve = [&](size_t bytes) -> void* {
    void* p = wb + off;
    off += (bytes + 255) & ~(size_t)255;
    return p;
  };
  unsigned short* xbf   = (unsigned short*)carve((size_t)N_NODES * 128 * 2);
  unsigned short* Wt[2] = {(unsigned short*)carve(896 * 128 * 2),
                           (unsigned short*)carve(896 * 128 * 2)};
  float* bias[2] = {(float*)carve(896 * 4), (float*)carve(896 * 4)};
  unsigned short* Wgt    = (unsigned short*)carve(128 * 256 * 2);
  unsigned short* W1t[2] = {(unsigned short*)carve(128 * 128 * 2),
                            (unsigned short*)carve(128 * 128 * 2)};
  float* qkvs = (float*)carve((size_t)N_NODES * QKV_W * 4);
  float* expv = (float*)carve((size_t)N_EDGES * 2 * 4);
  float* ssum = (float*)carve((size_t)N_NODES * 2 * 4);
  float* agg  = (float*)carve((size_t)N_NODES * 256 * 4);
  float* uB   = (float*)carve((size_t)N_NODES * 128 * 4);
  unsigned short* ubf = (unsigned short*)carve((size_t)N_NODES * 128 * 2);
  float* zB[2] = {(float*)carve((size_t)N_NODES * 128 * 4),
                  (float*)carve((size_t)N_NODES * 128 * 4)};
  unsigned short* zbf[2] = {(unsigned short*)carve((size_t)N_NODES * 128 * 2),
                            (unsigned short*)carve((size_t)N_NODES * 128 * 2)};
  float* wsum = (float*)carve(4 * 4);
  float* coef = (float*)carve(2 * 4);

  const long NE4 = (long)N_NODES * 32;               // 4 channels per thread
  const int EW_BLOCKS = (int)((NE4 + 255) / 256);

  // ---- prep: convert x, transpose-convert all weights, concat biases ----
  k_f32_to_bf16<<<EW_BLOCKS, 256, 0, stream>>>(features, xbf, NE4);
  for (int r = 0; r < 2; ++r) {
    k_transpose_cvt<<<(128 * 256 + 255) / 256, 256, 0, stream>>>(rW[r][0], Wt[r], 128, 256, 128, 0);
    k_transpose_cvt<<<(128 * 256 + 255) / 256, 256, 0, stream>>>(rW[r][1], Wt[r], 128, 256, 128, 256);
    k_transpose_cvt<<<(128 * 256 + 255) / 256, 256, 0, stream>>>(rW[r][2], Wt[r], 128, 256, 128, 512);
    k_transpose_cvt<<<(128 * 128 + 255) / 256, 256, 0, stream>>>(rW[r][3], Wt[r], 128, 128, 128, 768);
    hipMemcpyAsync(bias[r] + 0,   rb[r][0], 256 * 4, hipMemcpyDeviceToDevice, stream);
    hipMemcpyAsync(bias[r] + 256, rb[r][1], 256 * 4, hipMemcpyDeviceToDevice, stream);
    hipMemcpyAsync(bias[r] + 512, rb[r][2], 256 * 4, hipMemcpyDeviceToDevice, stream);
    hipMemcpyAsync(bias[r] + 768, rb[r][3], 128 * 4, hipMemcpyDeviceToDevice, stream);
  }
  k_transpose_cvt<<<(256 * 128 + 255) / 256, 256, 0, stream>>>(gate_W, Wgt, 256, 128, 256, 0);
  for (int h = 0; h < 2; ++h)
    k_transpose_cvt<<<(128 * 128 + 255) / 256, 256, 0, stream>>>(sW1[h], W1t[h], 128, 128, 128, 0);
  hipMemsetAsync(wsum, 0, 4 * sizeof(float), stream);

  // ---- per relation ----
  for (int r = 0; r < 2; ++r) {
    const int* src = ei[r];
    const int* dst = ei[r] + N_EDGES;
    k_gemm_qkvs<<<dim3(7, 3125), 256, 0, stream>>>(xbf, Wt[r], bias[r], qkvs);
    hipMemsetAsync(ssum, 0, (size_t)N_NODES * 2 * 4, stream);
    hipMemsetAsync(agg,  0, (size_t)N_NODES * 256 * 4, stream);
    k_edge_attn<<<N_EDGES / 8, 256, 0, stream>>>(src, dst, qkvs, expv, ssum);
    k_edge_scatter<<<N_EDGES / 8, 256, 0, stream>>>(src, dst, qkvs, expv, ssum, agg);
    k_make_u<<<EW_BLOCKS, 256, 0, stream>>>(agg, qkvs, uB, ubf);
    k_gate_fused<<<3125, 256, 0, stream>>>(ubf, xbf, Wgt, gate_b, uB, features,
                                           zB[r], zbf[r]);
  }

  // ---- semantic attention ----
  for (int h = 0; h < 2; ++h)
    for (int r = 0; r < 2; ++r)
      k_sem_reduce<<<3125, 256, 0, stream>>>(zbf[r], W1t[h], sb1[h], sw2[h],
                                             &wsum[h * 2 + r]);
  k_sem_coef<<<1, 32, 0, stream>>>(wsum, coef);
  k_final<<<EW_BLOCKS, 256, 0, stream>>>(zB[0], zB[1], coef, out);
}
